// VIT_83966610637440
// MI455X (gfx1250) — matrix operand order repaired and checked by reading, compiled, not run on hardware
//
#include <hip/hip_runtime.h>
#include <math.h>

// ---------------------------------------------------------------------------
// ViT forward for MI455X (gfx1250, wave32).
//  * all large matmuls: v_wmma_f32_16x16x32_bf16 (bf16 operands, f32 accum)
//  * tile feeds: global_load_async_to_lds_b128 (ASYNCcnt), double-buffered LDS
//  * every GEMM B operand is stored N x K (K contiguous), so A and B tile
//    copies are identical guarded row copies -> both go through the async path
// Dims: B=16 C=3 IMG=224 P=16 E=768 D=769 S=197 H=8 HID=3072 L=6 NCLS=1000
// Padded K dims (multiples of 32): D->800, D*H=6152->6176, S->224.
// ---------------------------------------------------------------------------

typedef unsigned short u16;
typedef __bf16 bf16_t;
typedef bf16_t v16bf __attribute__((ext_vector_type(16)));
typedef float  v8f   __attribute__((ext_vector_type(8)));

static __device__ __forceinline__ u16 f2bf(float f) {
  union { float f; unsigned u; } c; c.f = f;
  unsigned u = c.u;
  unsigned r = u + 0x7FFFu + ((u >> 16) & 1u);  // round-to-nearest-even
  return (u16)(r >> 16);
}

union Frag {
  u16   u[16];
  uint4 q[2];
  v16bf v;
};

struct GemmArgs {
  const u16* A; long asb; int lda;   // A: M x Kp row-major (K contiguous)
  const u16* B; long bsb; int ldb;   // B: N x K row-major (K contiguous)
  int M, N, Kp;                      // Kp multiple of 32
  int mode; float alpha;
  const float* bias;                 // indexed by global n (modes 0,1,2,4)
  float* of; long ldo; long ofb;     // f32 out (modes 0,1)
  u16* ob;                           // bf16 out (modes 2,3,4)
  long p0, p1;                       // scatter strides
};

// mode 0: of[z*ofb + m*ldo + n]  = alpha*acc + bias[n]          (f32)
// mode 1: of[m*ldo + n]         += acc + bias[n]                (f32 residual)
// mode 2: qkv head split: b=m/197,s=m%197,d=n/8,hh=n%8
//         ob[(b*8+hh)*p0 + s*p1 + d] = bf16(acc + bias[n])
// mode 3: attn-out concat: z=bh: b=z/8,hh=z%8
//         ob[(b*197+m)*p0 + hh*769 + n] = bf16(acc)
// mode 4: V head split, transposed (Vt[bh][d][s]):
//         ob[(b*8+hh)*p0 + d*p1 + s] = bf16(acc + bias[n])

__global__ __launch_bounds__(256) void gemm_bf16(GemmArgs g) {
  __shared__ u16 Al[2][128 * 40];   // 40-ushort row stride (bank friendly)
  __shared__ u16 Bl[2][128 * 40];   // B tile rows are output columns (N x K)

  const int tid  = threadIdx.x;
  const int zb   = blockIdx.z;
  const long m0  = (long)blockIdx.y * 128;
  const long n0  = (long)blockIdx.x * 128;
  const u16* A  = g.A + (long)zb * g.asb;
  const u16* Bm = g.B + (long)zb * g.bsb;

  const int lane = tid & 31, wid = tid >> 5;
  const int half = lane >> 4, l = lane & 15;
  const int wm = (wid & 1) * 64;   // 2 waves x 4 tiles x 16 = 128 rows
  const int wn = (wid >> 1) * 32;  // 4 waves x 2 tiles x 16 = 128 cols

  // copy assignment: 2 threads per row, 16 ushorts (32B) each
  const int crow = tid >> 1, ccol = (tid & 1) * 16;
  const long mrow = m0 + crow, nrow = n0 + crow;
  const bool mok = mrow < g.M, nok = nrow < g.N;
  const u16* Asrc = A + mrow * (long)g.lda + ccol;
  const u16* Bsrc = Bm + nrow * (long)g.ldb + ccol;

  const v8f zinit = {0.f,0.f,0.f,0.f,0.f,0.f,0.f,0.f};
  v8f acc[4][2];
#pragma unroll
  for (int a = 0; a < 4; ++a)
#pragma unroll
    for (int b = 0; b < 2; ++b) acc[a][b] = zinit;

  // async memory->LDS copy of one 128x32 double-tile (A and B), guarded rows
  // fall back to plain zero ds_stores (ordered by the __syncthreads below).
  auto issue = [&](int buf, int k0) {
    unsigned la = (unsigned)(unsigned long long)&Al[buf][crow * 40 + ccol];
    unsigned lb = (unsigned)(unsigned long long)&Bl[buf][crow * 40 + ccol];
    if (mok) {
      unsigned long long ga = (unsigned long long)(Asrc + k0);
      asm volatile("global_load_async_to_lds_b128 %0, %1, off\n\t"
                   "global_load_async_to_lds_b128 %0, %1, off offset:16"
                   :: "v"(la), "v"(ga) : "memory");
    } else {
      uint4 z = make_uint4(0, 0, 0, 0);
      uint4* d = reinterpret_cast<uint4*>(&Al[buf][crow * 40 + ccol]);
      d[0] = z; d[1] = z;
    }
    if (nok) {
      unsigned long long gb = (unsigned long long)(Bsrc + k0);
      asm volatile("global_load_async_to_lds_b128 %0, %1, off\n\t"
                   "global_load_async_to_lds_b128 %0, %1, off offset:16"
                   :: "v"(lb), "v"(gb) : "memory");
    } else {
      uint4 z = make_uint4(0, 0, 0, 0);
      uint4* d = reinterpret_cast<uint4*>(&Bl[buf][crow * 40 + ccol]);
      d[0] = z; d[1] = z;
    }
  };

  const int nk = g.Kp >> 5;
  issue(0, 0);
  for (int t = 0; t < nk; ++t) {
    const int buf = t & 1;
    asm volatile("s_wait_asynccnt 0x0" ::: "memory");  // tile t landed in LDS
    __syncthreads();                                   // ...for every wave
    if (t + 1 < nk) issue(buf ^ 1, (t + 1) << 5);      // overlap next copy

    // fragments in contiguous k order (k = 16*half + s) for A and B, then 8 WMMAs on the current buffer
    Frag bfr[2];
#pragma unroll
    for (int nt = 0; nt < 2; ++nt) {
      const uint4* p = reinterpret_cast<const uint4*>(
          &Bl[buf][(wn + nt * 16 + l) * 40 + half * 16]);
      bfr[nt].q[0] = p[0]; bfr[nt].q[1] = p[1];
    }
#pragma unroll
    for (int mt = 0; mt < 4; ++mt) {
      Frag af;
      const u16* base = &Al[buf][(wm + mt * 16 + l) * 40];
      af.q[0] = *reinterpret_cast<const uint4*>(base + half * 16);
      af.q[1] = *reinterpret_cast<const uint4*>(base + half * 16 + 8);
#pragma unroll
      for (int nt = 0; nt < 2; ++nt) {
        acc[mt][nt] = __builtin_amdgcn_wmma_f32_16x16x32_bf16(
            false, af.v, false, bfr[nt].v, (short)0, acc[mt][nt], false, false);
      }
    }
    // no trailing barrier needed: a wave only issues copies for tile t+1
    // after barrier t, by which point all waves consumed tile t-1's buffer.
  }

  // ---- epilogue (C/D layout: row = r + half*8, col = lane%16) ----
  if (g.mode == 0) {
    float* of = g.of + (long)zb * g.ofb;
#pragma unroll
    for (int mt = 0; mt < 4; ++mt)
#pragma unroll
      for (int nt = 0; nt < 2; ++nt) {
        long n = n0 + wn + nt * 16 + l;
        if (n >= g.N) continue;
        float bb = g.bias ? g.bias[n] : 0.f;
#pragma unroll
        for (int r = 0; r < 8; ++r) {
          long m = m0 + wm + mt * 16 + half * 8 + r;
          if (m >= g.M) continue;
          of[m * g.ldo + n] = g.alpha * acc[mt][nt][r] + bb;
        }
      }
  } else if (g.mode == 1) {
#pragma unroll
    for (int mt = 0; mt < 4; ++mt)
#pragma unroll
      for (int nt = 0; nt < 2; ++nt) {
        long n = n0 + wn + nt * 16 + l;
        if (n >= g.N) continue;
        float bb = g.bias[n];
#pragma unroll
        for (int r = 0; r < 8; ++r) {
          long m = m0 + wm + mt * 16 + half * 8 + r;
          if (m >= g.M) continue;
          g.of[m * g.ldo + n] += acc[mt][nt][r] + bb;
        }
      }
  } else if (g.mode == 2) {
#pragma unroll
    for (int mt = 0; mt < 4; ++mt)
#pragma unroll
      for (int nt = 0; nt < 2; ++nt) {
        long n = n0 + wn + nt * 16 + l;
        if (n >= g.N) continue;
        float bb = g.bias[n];
        long d = n >> 3; int hh = (int)(n & 7);
#pragma unroll
        for (int r = 0; r < 8; ++r) {
          long m = m0 + wm + mt * 16 + half * 8 + r;
          if (m >= g.M) continue;
          long b = m / 197, s = m % 197;
          g.ob[(b * 8 + hh) * g.p0 + s * g.p1 + d] = f2bf(acc[mt][nt][r] + bb);
        }
      }
  } else if (g.mode == 4) {
#pragma unroll
    for (int mt = 0; mt < 4; ++mt)
#pragma unroll
      for (int nt = 0; nt < 2; ++nt) {
        long n = n0 + wn + nt * 16 + l;
        if (n >= g.N) continue;
        float bb = g.bias[n];
        long d = n >> 3; int hh = (int)(n & 7);
#pragma unroll
        for (int r = 0; r < 8; ++r) {
          long m = m0 + wm + mt * 16 + half * 8 + r;
          if (m >= g.M) continue;
          long b = m / 197, s = m % 197;
          g.ob[(b * 8 + hh) * g.p0 + d * g.p1 + s] = f2bf(acc[mt][nt][r] + bb);
        }
      }
  } else { // mode 3
    int b = zb >> 3, hh = zb & 7;
#pragma unroll
    for (int mt = 0; mt < 4; ++mt)
#pragma unroll
      for (int nt = 0; nt < 2; ++nt) {
        long n = n0 + wn + nt * 16 + l;
        if (n >= g.N) continue;
#pragma unroll
        for (int r = 0; r < 8; ++r) {
          long m = m0 + wm + mt * 16 + half * 8 + r;
          if (m >= g.M) continue;
          g.ob[((long)(b * 197) + m) * g.p0 + (long)hh * 769 + n] = f2bf(acc[mt][nt][r]);
        }
      }
  }
}

// --------------------------- support kernels -------------------------------

__global__ void zero_u4(void* p, size_t n16) {
  size_t i = (size_t)blockIdx.x * blockDim.x + threadIdx.x;
  size_t st = (size_t)gridDim.x * blockDim.x;
  uint4* q = (uint4*)p;
  uint4 z = make_uint4(0, 0, 0, 0);
  for (; i < n16; i += st) q[i] = z;
}

// f32 (K x N, row-major) -> bf16 transposed (N x Kp, K contiguous, zero pad).
// 32x32 LDS tile: coalesced reads AND coalesced transposed writes.
__global__ __launch_bounds__(256) void cvt_t(const float* __restrict__ src,
                                             u16* __restrict__ dst,
                                             int K, int N, int Kp) {
  __shared__ u16 t[32][33];
  const int kb = blockIdx.x * 32;
  const int nb = blockIdx.y * 32;
  const int r = threadIdx.x >> 5;   // 0..7
  const int c = threadIdx.x & 31;   // 0..31
#pragma unroll
  for (int p = 0; p < 4; ++p) {
    int k = kb + p * 8 + r;
    int n = nb + c;
    float v = (k < K && n < N) ? src[(long)k * N + n] : 0.f;
    t[p * 8 + r][c] = f2bf(v);
  }
  __syncthreads();
#pragma unroll
  for (int p = 0; p < 4; ++p) {
    int n = nb + p * 8 + r;
    if (n < N) dst[(long)n * Kp + kb + c] = t[c][p * 8 + r];
  }
}

__global__ void init_h(float* h) {
  int idx = blockIdx.x * blockDim.x + threadIdx.x;
  if (idx < 16 * 197) {                 // position channel: h[b,s,0] = s
    int b = idx / 197, s = idx % 197;
    h[((long)(b * 197 + s)) * 769] = (float)s;
  }
  int j = idx - 16 * 197;               // cls token row: h[b,0,1+e] = 1
  if (j >= 0 && j < 16 * 768) {
    int b = j / 768, e = j % 768;
    h[((long)(b * 197)) * 769 + 1 + e] = 1.0f;
  }
}

__global__ __launch_bounds__(256) void patch_embed(
    const float* __restrict__ x, const float* __restrict__ w,
    const float* __restrict__ bias, float* __restrict__ h) {
  long idx = (long)blockIdx.x * blockDim.x + threadIdx.x;
  if (idx >= (long)16 * 196 * 768) return;
  int e = (int)(idx % 768);
  int p = (int)((idx / 768) % 196);
  int b = (int)(idx / (768 * 196));
  int py = p / 14, px = p % 14;
  float acc = bias[e];
  for (int c = 0; c < 3; ++c) {
    const float* wrow  = w + (long)(e * 3 + c) * 256;
    const float* xbase = x + (((long)(b * 3 + c) * 224 + py * 16) * 224 + px * 16);
#pragma unroll 4
    for (int i = 0; i < 16; ++i) {
      const float4* xr = (const float4*)(xbase + (long)i * 224);
      const float4* wr = (const float4*)(wrow + i * 16);
#pragma unroll
      for (int j = 0; j < 4; ++j) {
        float4 xv = xr[j], wv = wr[j];
        acc += xv.x * wv.x + xv.y * wv.y + xv.z * wv.z + xv.w * wv.w;
      }
    }
  }
  h[((long)(b * 197) + 1 + p) * 769 + 1 + e] = acc;
}

__global__ __launch_bounds__(256) void ln_rows(
    const float* __restrict__ h, const float* __restrict__ g,
    const float* __restrict__ b, u16* __restrict__ z, int D_, int ldz) {
  int m = blockIdx.x, tid = threadIdx.x;
  const float* row = h + (long)m * D_;
  __shared__ float sh[256];
  float s = 0.f, s2 = 0.f;
  for (int c = tid; c < D_; c += 256) { float x = row[c]; s += x; s2 += x * x; }
  sh[tid] = s; __syncthreads();
  for (int o = 128; o > 0; o >>= 1) { if (tid < o) sh[tid] += sh[tid + o]; __syncthreads(); }
  float mean = sh[0] / D_;
  __syncthreads();
  sh[tid] = s2; __syncthreads();
  for (int o = 128; o > 0; o >>= 1) { if (tid < o) sh[tid] += sh[tid + o]; __syncthreads(); }
  float var = sh[0] / D_ - mean * mean;
  float rstd = rsqrtf(var + 1e-6f);
  u16* zr = z + (long)m * ldz;
  for (int c = tid; c < D_; c += 256)
    zr[c] = f2bf((row[c] - mean) * rstd * g[c] + b[c]);
}

// reference softmaxes over axis=2 (the key index i) for each query j.
__global__ __launch_bounds__(256) void softmax_col(
    const float* __restrict__ Sf, u16* __restrict__ Sb,
    int S_, int ld, long bsF, long bsB) {
  int zb = blockIdx.x, j = threadIdx.x;
  if (j >= S_) return;
  const float* p = Sf + (long)zb * bsF + j;
  float mx = -3.0e38f;
  for (int i = 0; i < S_; ++i) mx = fmaxf(mx, p[(long)i * ld]);
  float sum = 0.f;
  for (int i = 0; i < S_; ++i) sum += expf(p[(long)i * ld] - mx);
  float inv = 1.f / sum;
  u16* q = Sb + (long)zb * bsB + j;
  for (int i = 0; i < S_; ++i) q[(long)i * ld] = f2bf(expf(p[(long)i * ld] - mx) * inv);
}

// batchnorm over axes (batch, channel) per sequence position s, + exact GELU.
__global__ __launch_bounds__(256) void bn_gelu(
    const float* __restrict__ Y, const float* __restrict__ g,
    const float* __restrict__ b, u16* __restrict__ O, int S_, int C) {
  int s = blockIdx.x, tid = threadIdx.x;
  __shared__ float sh[256];
  long n = (long)16 * C;
  float sum = 0.f, sum2 = 0.f;
  for (long t = tid; t < n; t += 256) {
    long bb = t / C, c = t % C;
    float x = Y[(bb * S_ + s) * (long)C + c];
    sum += x; sum2 += x * x;
  }
  sh[tid] = sum; __syncthreads();
  for (int o = 128; o > 0; o >>= 1) { if (tid < o) sh[tid] += sh[tid + o]; __syncthreads(); }
  float mean = sh[0] / n;
  __syncthreads();
  sh[tid] = sum2; __syncthreads();
  for (int o = 128; o > 0; o >>= 1) { if (tid < o) sh[tid] += sh[tid + o]; __syncthreads(); }
  float var = sh[0] / n - mean * mean;
  float rstd = rsqrtf(var + 1e-5f);
  float gg = g[s], bb_ = b[s];
  for (long t = tid; t < n; t += 256) {
    long bb = t / C, c = t % C;
    long off = (bb * S_ + s) * (long)C + c;
    float v = (Y[off] - mean) * rstd * gg + bb_;
    O[off] = f2bf(0.5f * v * (1.f + erff(v * 0.70710678118654752f)));
  }
}

__global__ __launch_bounds__(256) void bn_res(
    const float* __restrict__ Y, const float* __restrict__ g,
    const float* __restrict__ b, float* __restrict__ h, int S_, int C) {
  int s = blockIdx.x, tid = threadIdx.x;
  __shared__ float sh[256];
  long n = (long)16 * C;
  float sum = 0.f, sum2 = 0.f;
  for (long t = tid; t < n; t += 256) {
    long bb = t / C, c = t % C;
    float x = Y[(bb * S_ + s) * (long)C + c];
    sum += x; sum2 += x * x;
  }
  sh[tid] = sum; __syncthreads();
  for (int o = 128; o > 0; o >>= 1) { if (tid < o) sh[tid] += sh[tid + o]; __syncthreads(); }
  float mean = sh[0] / n;
  __syncthreads();
  sh[tid] = sum2; __syncthreads();
  for (int o = 128; o > 0; o >>= 1) { if (tid < o) sh[tid] += sh[tid + o]; __syncthreads(); }
  float var = sh[0] / n - mean * mean;
  float rstd = rsqrtf(var + 1e-5f);
  float gg = g[s], bb_ = b[s];
  for (long t = tid; t < n; t += 256) {
    long bb = t / C, c = t % C;
    long off = (bb * S_ + s) * (long)C + c;
    h[off] += (Y[off] - mean) * rstd * gg + bb_;
  }
}

__global__ void head_k(const float* __restrict__ h, const float* __restrict__ w,
                       const float* __restrict__ bias, float* __restrict__ out) {
  int idx = blockIdx.x * blockDim.x + threadIdx.x;
  if (idx >= 16 * 1000) return;
  int n = idx % 1000, b = idx / 1000;
  const float* row = h + (long)b * 197 * 769;  // cls token (s == 0)
  float s = bias[n];
  for (int c = 0; c < 769; ++c) s += row[c] * w[(long)c * 1000 + n];
  out[idx] = s;
}

// ------------------------------ launcher -----------------------------------

extern "C" void kernel_launch(void* const* d_in, const int* in_sizes, int n_in,
                              void* d_out, int out_size, void* d_ws, size_t ws_size,
                              hipStream_t stream) {
  (void)in_sizes; (void)n_in; (void)out_size; (void)ws_size;
  const float* x      = (const float*)d_in[0];
  const float* conv_w = (const float*)d_in[1];
  const float* conv_b = (const float*)d_in[2];
  const float* ln_g   = (const float*)d_in[3];
  const float* ln_b   = (const float*)d_in[4];
  const float* wq     = (const float*)d_in[5];
  const float* bq     = (const float*)d_in[6];
  const float* wk     = (const float*)d_in[7];
  const float* bk     = (const float*)d_in[8];
  const float* wv     = (const float*)d_in[9];
  const float* bv     = (const float*)d_in[10];
  const float* wo     = (const float*)d_in[11];
  const float* bo     = (const float*)d_in[12];
  const float* fc1w   = (const float*)d_in[13];
  const float* fc1bi  = (const float*)d_in[14];
  const float* fc2w   = (const float*)d_in[15];
  const float* fc2bi  = (const float*)d_in[16];
  const float* bn1g   = (const float*)d_in[17];
  const float* bn1b   = (const float*)d_in[18];
  const float* bn2g   = (const float*)d_in[19];
  const float* bn2b   = (const float*)d_in[20];
  const float* headw  = (const float*)d_in[21];
  const float* headb  = (const float*)d_in[22];
  float* out = (float*)d_out;

  const int Bv = 16, S = 197, D = 769, H = 8, DH = 6152, HID = 3072, L = 6;
  const int M  = Bv * S;     // 3152
  const int DP = 800;        // D  padded to /32
  const int DHP = 6176;      // DH padded to /32
  const int SK = 224;        // S  padded to /32 (K of the A@V GEMM)
  const int BH = Bv * H;     // 128

  char* ws = (char*)d_ws;
  size_t off = 0;
  auto alloc = [&](size_t bytes) -> char* {
    char* p = ws + off;
    off += (bytes + 255) & ~(size_t)255;
    return p;
  };
  float* h   = (float*)alloc((size_t)M * D * 4);
  float* Y1  = (float*)alloc((size_t)M * HID * 4);
  float* Y2  = (float*)alloc((size_t)M * D * 4);
  float* Sf  = (float*)alloc((size_t)BH * S * SK * 4);
  char*  z0  = ws + off;     // start of zero-padded bf16 arena
  u16* Zb  = (u16*)alloc((size_t)M * DP * 2);
  u16* Qh  = (u16*)alloc((size_t)BH * S * DP * 2);
  u16* Kh  = (u16*)alloc((size_t)BH * S * DP * 2);
  u16* Vt  = (u16*)alloc((size_t)BH * D * SK * 2);   // transposed: [bh][d][s]
  u16* Sb  = (u16*)alloc((size_t)BH * S * SK * 2);
  u16* Ob  = (u16*)alloc((size_t)M * DHP * 2);
  size_t zbytes = (size_t)((ws + off) - z0);
  u16* Y1b = (u16*)alloc((size_t)M * HID * 2);
  u16* wqb = (u16*)alloc((size_t)DH * DP * 2);       // N x K layouts
  u16* wkb = (u16*)alloc((size_t)DH * DP * 2);
  u16* wvb = (u16*)alloc((size_t)DH * DP * 2);
  u16* wob = (u16*)alloc((size_t)D * DHP * 2);
  u16* f1b = (u16*)alloc((size_t)HID * DP * 2);
  u16* f2b = (u16*)alloc((size_t)D * HID * 2);

  auto gemm = [&](const u16* A, long asb, int lda,
                  const u16* Bm, long bsb, int ldb,
                  int Mm, int Nn, int Kp, int batches,
                  int mode, float alpha, const float* bias,
                  float* of, long ldo, long ofb,
                  u16* ob, long p0, long p1) {
    GemmArgs g;
    g.A = A; g.asb = asb; g.lda = lda;
    g.B = Bm; g.bsb = bsb; g.ldb = ldb;
    g.M = Mm; g.N = Nn; g.Kp = Kp;
    g.mode = mode; g.alpha = alpha; g.bias = bias;
    g.of = of; g.ldo = ldo; g.ofb = ofb;
    g.ob = ob; g.p0 = p0; g.p1 = p1;
    dim3 grid((Nn + 127) / 128, (Mm + 127) / 128, batches);
    gemm_bf16<<<grid, 256, 0, stream>>>(g);
  };

  // Zero padded bf16 arena so K-padding contributes exactly 0 to every WMMA.
  zero_u4<<<2048, 256, 0, stream>>>(z0, zbytes / 16);

  // Embedding: position channel + cls row + patch conv (direct, ~0.1% FLOPs).
  init_h<<<(16 * 197 + 16 * 768 + 255) / 256, 256, 0, stream>>>(h);
  patch_embed<<<(int)(((long)16 * 196 * 768 + 255) / 256), 256, 0, stream>>>(x, conv_w, conv_b, h);

  const float scale = (float)(1.0 / sqrt((double)D));

  for (int i = 0; i < L; ++i) {
    // f32 -> bf16 weight conversion, transposed to N x K, zero K-padding
    cvt_t<<<dim3(DP / 32, (DH + 31) / 32), 256, 0, stream>>>(wq + (long)i * D * DH, wqb, D, DH, DP);
    cvt_t<<<dim3(DP / 32, (DH + 31) / 32), 256, 0, stream>>>(wk + (long)i * D * DH, wkb, D, DH, DP);
    cvt_t<<<dim3(DP / 32, (DH + 31) / 32), 256, 0, stream>>>(wv + (long)i * D * DH, wvb, D, DH, DP);
    cvt_t<<<dim3(DHP / 32, (D + 31) / 32), 256, 0, stream>>>(wo + (long)i * DH * D, wob, DH, D, DHP);
    cvt_t<<<dim3(DP / 32, (HID + 31) / 32), 256, 0, stream>>>(fc1w + (long)i * D * HID, f1b, D, HID, DP);
    cvt_t<<<dim3(HID / 32, (D + 31) / 32), 256, 0, stream>>>(fc2w + (long)i * HID * D, f2b, HID, D, HID);

    // z = LN(h) -> bf16 (padded cols stay zero)
    ln_rows<<<M, 256, 0, stream>>>(h, ln_g + (long)i * D, ln_b + (long)i * D, Zb, D, DP);

    // q,k,v projections; epilogue scatters into per-head layouts
    gemm(Zb, 0, DP, wqb, 0, DP, M, DH, DP, 1, 2, 1.f, bq + (long)i * DH,
         nullptr, 0, 0, Qh, (long)S * DP, DP);
    gemm(Zb, 0, DP, wkb, 0, DP, M, DH, DP, 1, 2, 1.f, bk + (long)i * DH,
         nullptr, 0, 0, Kh, (long)S * DP, DP);
    gemm(Zb, 0, DP, wvb, 0, DP, M, DH, DP, 1, 4, 1.f, bv + (long)i * DH,
         nullptr, 0, 0, Vt, (long)D * SK, SK);

    // scores[bh] = K · Qᵀ  (A[i,j] = Σ_d k[i,d] q[j,d]), scaled by D^-0.5
    gemm(Kh, (long)S * DP, DP, Qh, (long)S * DP, DP, S, S, DP, BH,
         0, scale, nullptr, Sf, SK, (long)S * SK, nullptr, 0, 0);
    softmax_col<<<BH, 256, 0, stream>>>(Sf, Sb, S, SK, (long)S * SK, (long)S * SK);

    // o[bh] = P · V (B operand is Vt = N x K), scattered into concat layout
    gemm(Sb, (long)S * SK, SK, Vt, (long)D * SK, SK, S, D, SK, BH,
         3, 1.f, nullptr, nullptr, 0, 0, Ob, DHP, 0);

    // h += o @ wo + bo
    gemm(Ob, 0, DHP, wob, 0, DHP, M, D, DHP, 1,
         1, 1.f, bo + (long)i * D, h, D, 0, nullptr, 0, 0);

    // MLP with per-position batchnorms
    ln_rows<<<M, 256, 0, stream>>>(h, ln_g + (long)i * D, ln_b + (long)i * D, Zb, D, DP);
    gemm(Zb, 0, DP, f1b, 0, DP, M, HID, DP, 1,
         0, 1.f, fc1bi + (long)i * HID, Y1, HID, 0, nullptr, 0, 0);
    bn_gelu<<<S, 256, 0, stream>>>(Y1, bn1g + (long)i * S, bn1b + (long)i * S, Y1b, S, HID);
    gemm(Y1b, 0, HID, f2b, 0, HID, M, D, HID, 1,
         0, 1.f, fc2bi + (long)i * D, Y2, D, 0, nullptr, 0, 0);
    bn_res<<<S, 256, 0, stream>>>(Y2, bn2g + (long)i * S, bn2b + (long)i * S, h, S, D);
  }

  head_k<<<(16 * 1000 + 255) / 256, 256, 0, stream>>>(h, headw, headb, out);
}
